// MHSAModule_36893769073002
// MI455X (gfx1250) — compile-verified, matrix operand order checked by reading, not run on hardware
//
#include <hip/hip_runtime.h>
#include <hip/hip_bf16.h>

// MHSA (Conformer / Transformer-XL relative attention) for gfx1250.
// B=16, T=1024, DM=256, H=4, HS=64. All GEMMs via v_wmma_f32_16x16x32_f16.
// K tiles staged to LDS with the Tensor Data Mover (tensor_load_to_lds).

typedef __attribute__((ext_vector_type(16))) _Float16 v16h;
typedef __attribute__((ext_vector_type(8)))  _Float16 v8h;
typedef __attribute__((ext_vector_type(8)))  float    v8f;
typedef __attribute__((ext_vector_type(4)))  unsigned int u32x4;
typedef __attribute__((ext_vector_type(8)))  int      i32x8;
typedef __attribute__((ext_vector_type(4)))  int      i32x4;

#define B_  16
#define T_  1024
#define DM_ 256
#define H_  4
#define HS_ 64

__device__ inline v8f wmma32(v16h a, v16h b, v8f c) {
  return __builtin_amdgcn_wmma_f32_16x16x32_f16(
      /*neg_a=*/false, a, /*neg_b=*/false, b,
      /*c_mod=*/(short)0, c, /*reuse_a=*/false, /*reuse_b=*/false);
}

__device__ inline v8f zero8() {
  v8f z = {0.f, 0.f, 0.f, 0.f, 0.f, 0.f, 0.f, 0.f};
  return z;
}

// A-fragment (16x32 f16, M rows x K cols), or B-fragment when the B source is
// stored so that the fragment's N-column is the row index of the buffer
// (e.g. K[T][HS] for S=Q*K^T, or Vt[HS][T] for attn@V).
// Lane layout (ISA 7.12.2): m = lane&15; K chunks at 8*(lane>>4) and +16.
__device__ inline v16h frag_rows(const _Float16* base, int row0, int ld,
                                 int kOff, int lane) {
  int m = lane & 15, hf = lane >> 4;
  const _Float16* p = base + (size_t)(row0 + m) * ld + kOff + 8 * hf;
  v8h lo = *(const v8h*)p;
  v8h hi = *(const v8h*)(p + 16);
  v16h r;
#pragma unroll
  for (int i = 0; i < 8; ++i) { r[i] = lo[i]; r[i + 8] = hi[i]; }
  return r;
}

__device__ inline v16h frag_rows_clamped(const _Float16* base, int row0,
                                         int rmax, int ld, int kOff, int lane) {
  int m = lane & 15, hf = lane >> 4;
  int r = row0 + m;
  r = r < 0 ? 0 : (r > rmax - 1 ? rmax - 1 : r);
  const _Float16* p = base + (size_t)r * ld + kOff + 8 * hf;
  v8h lo = *(const v8h*)p;
  v8h hi = *(const v8h*)(p + 16);
  v16h q;
#pragma unroll
  for (int i = 0; i < 8; ++i) { q[i] = lo[i]; q[i + 8] = hi[i]; }
  return q;
}

__device__ inline float bperm(float v, int srclane) {
  return __int_as_float(
      __builtin_amdgcn_ds_bpermute(srclane << 2, __float_as_int(v)));
}

__device__ inline float redmax16(float x) {
  x = fmaxf(x, __shfl_xor(x, 1, 32));
  x = fmaxf(x, __shfl_xor(x, 2, 32));
  x = fmaxf(x, __shfl_xor(x, 4, 32));
  x = fmaxf(x, __shfl_xor(x, 8, 32));
  return x;
}
__device__ inline float redsum16(float x) {
  x += __shfl_xor(x, 1, 32);
  x += __shfl_xor(x, 2, 32);
  x += __shfl_xor(x, 4, 32);
  x += __shfl_xor(x, 8, 32);
  return x;
}

// Tensor Data Mover: DMA a 2D row-major tile (tile_d0 x tile_d1 elements of
// 2 bytes, row stride = stride0 elements) from global to LDS. Tracked with
// TENSORcnt. Descriptor layout per CDNA5 ISA ch.8 (D# groups 0/1).
__device__ inline void tdm_load_2d_f16(unsigned int lds_off, const void* gaddr,
                                       unsigned int tensor_d0,
                                       unsigned int tensor_d1,
                                       unsigned int tile_d0,
                                       unsigned int tile_d1,
                                       unsigned int stride0) {
  unsigned long long ga = (unsigned long long)(size_t)gaddr;
  u32x4 g0;
  g0[0] = 1u;                                  // count=1, user-mode, no gather
  g0[1] = lds_off;                             // lds_addr (bytes)
  g0[2] = (unsigned int)ga;                    // global_addr[31:0]
  g0[3] = (unsigned int)((ga >> 32) & 0x01FFFFFFull) | (2u << 30); // addr[56:32], type=2
  i32x8 g1;
  g1[0] = 0x00010000;                          // data_size = 2 bytes
  g1[1] = (int)((tensor_d0 & 0xFFFFu) << 16);  // tensor_dim0[15:0]
  g1[2] = (int)((tensor_d0 >> 16) | ((tensor_d1 & 0xFFFFu) << 16));
  g1[3] = (int)((tensor_d1 >> 16) | (tile_d0 << 16));   // tile_dim0
  g1[4] = (int)(tile_d1 & 0xFFFFu);            // tile_dim1 (tile_dim2 = 0)
  g1[5] = (int)stride0;                        // tensor_dim0_stride[31:0]
  g1[6] = 0;                                   // stride hi / dim1 stride
  g1[7] = 0;
#if __has_builtin(__builtin_amdgcn_tensor_load_to_lds)
  i32x4 z4 = {0, 0, 0, 0};
  i32x8 z8 = {0, 0, 0, 0, 0, 0, 0, 0};
  __builtin_amdgcn_tensor_load_to_lds(g0, g1, z4, z4, z8, 0);
#else
  asm volatile("tensor_load_to_lds %0, %1" :: "s"(g0), "s"(g1) : "memory");
#endif
}

// ---------------- Kernel 1: LayerNorm -> f16 ----------------
__global__ __launch_bounds__(256) void ln_kernel(const float* __restrict__ x,
                                                 const float* __restrict__ g,
                                                 const float* __restrict__ be,
                                                 _Float16* __restrict__ out) {
  int row = blockIdx.x, t = threadIdx.x;
  float v = x[(size_t)row * DM_ + t];
  __shared__ float s1[256], s2[256];
  s1[t] = v; s2[t] = v * v;
  __syncthreads();
#pragma unroll
  for (int off = 128; off > 0; off >>= 1) {
    if (t < off) { s1[t] += s1[t + off]; s2[t] += s2[t + off]; }
    __syncthreads();
  }
  float mu  = s1[0] * (1.f / DM_);
  float var = s2[0] * (1.f / DM_) - mu * mu;
  float y = (v - mu) * rsqrtf(var + 1e-3f) * g[t] + be[t];
  out[(size_t)row * DM_ + t] = (_Float16)y;
}

// ---------------- Kernel 2: f32 -> f16 convert (pos_enc) ----------------
__global__ void cvt_kernel(const float* __restrict__ in,
                           _Float16* __restrict__ out, int n) {
  for (int i = blockIdx.x * blockDim.x + threadIdx.x; i < n;
       i += gridDim.x * blockDim.x)
    out[i] = (_Float16)in[i];
}

// ---------------- Kernel 3: transpose + convert a 256x256 weight ----------
__global__ __launch_bounds__(256) void tr_kernel(const float* __restrict__ W,
                                                 _Float16* __restrict__ Wt) {
  int d = blockIdx.x, c = threadIdx.x;
  Wt[(size_t)c * DM_ + d] = (_Float16)W[(size_t)d * DM_ + c];
}

// ---------------- Kernel 4: fused q/k/v/p projections (WMMA) -------------
// One wave = one 16x16 output tile; K loop over DM=256 in 8 steps of 32.
// V is written transposed (Vt[B,H,HS,T]) so attn@V B-fragments are contiguous.
__global__ __launch_bounds__(256) void proj_kernel(
    const _Float16* __restrict__ xh, const _Float16* __restrict__ ph,
    const _Float16* __restrict__ WqT, const _Float16* __restrict__ WkT,
    const _Float16* __restrict__ WvT, const _Float16* __restrict__ WpT,
    const float* __restrict__ bq, const float* __restrict__ bk,
    const float* __restrict__ bv, const float* __restrict__ pbu,
    const float* __restrict__ pbv, _Float16* __restrict__ qu,
    _Float16* __restrict__ qv, _Float16* __restrict__ kh,
    _Float16* __restrict__ vt, _Float16* __restrict__ pph) {
  int lane = threadIdx.x & 31, wid = threadIdx.x >> 5;
  int gid = blockIdx.x * 8 + wid;       // 65536 waves total
  int ct  = gid & 15;
  int rt  = (gid >> 4) & 63;
  int b   = (gid >> 10) & 15;
  int mat = gid >> 14;                  // 0=Q 1=K 2=V 3=P

  const _Float16* A  = (mat == 3) ? ph : xh;
  const _Float16* Wt = (mat == 0) ? WqT : (mat == 1) ? WkT
                      : (mat == 2) ? WvT : WpT;
  int r0 = b * T_ + rt * 16;
  int c0 = ct * 16;

  v8f acc = zero8();
#pragma unroll
  for (int kk = 0; kk < 8; ++kk) {
    v16h a = frag_rows(A, r0, DM_, kk * 32, lane);
    v16h w = frag_rows(Wt, c0, DM_, kk * 32, lane);   // B[k][n] = Wt[n][k] = W[k][n]
    acc = wmma32(a, w, acc);
  }

  int nl = lane & 15, hf = lane >> 4;
#pragma unroll
  for (int v = 0; v < 8; ++v) {
    int t  = rt * 16 + v + 8 * hf;
    int c  = c0 + nl;
    int hh = c >> 6, s = c & 63;
    size_t oidx = (((size_t)b * H_ + hh) * T_ + t) * HS_ + s;
    float a = acc[v];
    if (mat == 0) {
      float f = a + bq[c];
      qu[oidx] = (_Float16)(f + pbu[c]);
      qv[oidx] = (_Float16)(f + pbv[c]);
    } else if (mat == 1) {
      kh[oidx] = (_Float16)(a + bk[c]);
    } else if (mat == 2) {
      // transposed: Vt[b][h][s][t]
      vt[(((size_t)b * H_ + hh) * HS_ + s) * T_ + t] = (_Float16)(a + bv[c]);
    } else {
      pph[oidx] = (_Float16)a;
    }
  }
}

// ---------------- Kernel 5: fused rel-attention (flash style) ------------
// One wave owns a 16-row M-tile of one (b,h); loops over T in 32-col steps.
// K tile for each step is DMA'd to LDS with the Tensor Data Mover; the
// rel_shift is realized by WMMA G-panels + ds_bpermute diagonal gather.
__global__ __launch_bounds__(256) void attn_kernel(
    const _Float16* __restrict__ qu, const _Float16* __restrict__ qv,
    const _Float16* __restrict__ kk, const _Float16* __restrict__ vt,
    const _Float16* __restrict__ pp, _Float16* __restrict__ ao) {
  __shared__ alignas(16) _Float16 Kstage[8][32][HS_];  // 32 KB
  __shared__ alignas(16) _Float16 Pbuf[8][16][32];     // 8 KB
  int lane = threadIdx.x & 31, wid = threadIdx.x >> 5;
  int gid = blockIdx.x * 8 + wid;       // 4096 waves total
  int mt = gid & 63;
  int hh = (gid >> 6) & 3;
  int b  = gid >> 8;
  int m0 = mt * 16;
  size_t bh = ((size_t)b * H_ + hh) * T_ * HS_;
  const _Float16* qub = qu + bh;
  const _Float16* qvb = qv + bh;
  const _Float16* kb  = kk + bh;
  const _Float16* vtb = vt + bh;       // [HS][T]
  const _Float16* pb  = pp + bh;
  unsigned int kstage_off = (unsigned int)(size_t)&Kstage[wid][0][0];

  // Q fragments are reused across the whole column loop.
  v16h quA[2], qvA[2], qvB[2];
#pragma unroll
  for (int i = 0; i < 2; ++i) {
    quA[i] = frag_rows(qub, m0, HS_, i * 32, lane);
    qvA[i] = frag_rows(qvb, m0, HS_, i * 32, lane);
    qvB[i] = frag_rows_clamped(qvb, m0 + 1, T_, HS_, i * 32, lane); // rows m+1
  }

  float mrow[8], lrow[8];
  v8f o[4];
#pragma unroll
  for (int v = 0; v < 8; ++v) { mrow[v] = -1e30f; lrow[v] = 0.f; }
#pragma unroll
  for (int st = 0; st < 4; ++st) o[st] = zero8();

  int nl = lane & 15, hf = lane >> 4;
  const float inv_sqrt_hs = 0.125f;

#pragma unroll 1
  for (int step = 0; step < T_ / 32; ++step) {
    // TDM: stage this step's 32x64 K tile into LDS (overlaps with the
    // positional-panel global loads + WMMAs issued below).
    tdm_load_2d_f16(kstage_off, kb + (size_t)step * 32 * HS_,
                    /*tensor_d0=*/HS_, /*tensor_d1=*/32,
                    /*tile_d0=*/HS_, /*tile_d1=*/32, /*stride0=*/HS_);
    if (step + 1 < T_ / 32)  // warm L2 for next Vt panel
      __builtin_prefetch(vtb + (size_t)nl * T_ + (step + 1) * 32, 0, 1);

    v8f lg[2];
#pragma unroll
    for (int sub = 0; sub < 2; ++sub) {
      int n0 = step * 32 + sub * 16;

      // positional panels G[m, r] = qv[m] . p[r] (two row variants)
      int rA0 = n0 - m0 + (T_ - 16);   // for n <= m   : r = n-m+T-1
      int rB0 = n0 - m0 - 17;          // for n >= m+2 : r = n-m-2 (rows m+1)
      v8f gA0 = zero8(), gA1 = zero8(), gB0 = zero8(), gB1 = zero8();
      if (n0 <= m0 + 15) {             // some entry has n <= m
#pragma unroll
        for (int ks = 0; ks < 2; ++ks) {
          gA0 = wmma32(qvA[ks], frag_rows_clamped(pb, rA0,      T_, HS_, ks * 32, lane), gA0);
          gA1 = wmma32(qvA[ks], frag_rows_clamped(pb, rA0 + 16, T_, HS_, ks * 32, lane), gA1);
        }
      }
      if (n0 + 15 >= m0 + 2) {         // some entry has n >= m+2
#pragma unroll
        for (int ks = 0; ks < 2; ++ks) {
          gB0 = wmma32(qvB[ks], frag_rows_clamped(pb, rB0,      T_, HS_, ks * 32, lane), gB0);
          gB1 = wmma32(qvB[ks], frag_rows_clamped(pb, rB0 + 16, T_, HS_, ks * 32, lane), gB1);
        }
      }

      // content scores from the LDS-staged K tile: Qu (16xHS) x K^T
      if (sub == 0) __builtin_amdgcn_s_wait_tensorcnt(0);
      v8f s = zero8();
#pragma unroll
      for (int ks = 0; ks < 2; ++ks)
        s = wmma32(quA[ks],
                   frag_rows(&Kstage[wid][0][0], sub * 16, HS_, ks * 32, lane),
                   s);

      // rel-shift gather: element (m,n) needs panel column cl = nl-ml+15.
#pragma unroll
      for (int v = 0; v < 8; ++v) {
        int ml = v + 8 * hf;
        int cl = nl - ml + 15;                    // in [0,30]
        int sl = 16 * hf + (cl & 15);
        float aval = (cl < 16) ? bperm(gA0[v], sl) : bperm(gA1[v], sl);
        float bval = (cl < 16) ? bperm(gB0[v], sl) : bperm(gB1[v], sl);
        int dd = (n0 - m0) + nl - ml;             // n_abs - m_abs
        float pv = (dd <= 0) ? aval : ((dd == 1) ? 0.f : bval);
        lg[sub][v] = (s[v] + pv) * inv_sqrt_hs;
      }
    }

    // flash-softmax update over the 32 new columns
#pragma unroll
    for (int v = 0; v < 8; ++v) {
      int ml = v + 8 * hf;
      float mx = redmax16(fmaxf(lg[0][v], lg[1][v]));
      float mnew = fmaxf(mrow[v], mx);
      float corr = __expf(mrow[v] - mnew);
      mrow[v] = mnew;
      lrow[v] *= corr;
#pragma unroll
      for (int st = 0; st < 4; ++st) o[st][v] *= corr;
      float p0 = __expf(lg[0][v] - mnew);
      float p1 = __expf(lg[1][v] - mnew);
      lrow[v] += redsum16(p0 + p1);
      Pbuf[wid][ml][nl]      = (_Float16)p0;
      Pbuf[wid][ml][16 + nl] = (_Float16)p1;
    }
    asm volatile("s_wait_dscnt 0x0" ::: "memory");

    // out += probs(16x32) x V(32xHS); B-fragment from Vt rows (contiguous)
    v16h pa = frag_rows(&Pbuf[wid][0][0], 0, 32, 0, lane);
#pragma unroll
    for (int st = 0; st < 4; ++st)
      o[st] = wmma32(pa, frag_rows(vtb, st * 16, T_, step * 32, lane), o[st]);
  }

  // normalize and store attn_out as [B, T, H, HS] f16
#pragma unroll
  for (int st = 0; st < 4; ++st) {
#pragma unroll
    for (int v = 0; v < 8; ++v) {
      int m_abs = m0 + v + 8 * hf;
      int s_abs = st * 16 + nl;
      float val = o[st][v] / lrow[v];
      ao[(((size_t)b * T_ + m_abs) * H_ + hh) * HS_ + s_abs] = (_Float16)val;
    }
  }
}

// ---------------- Kernel 6: output projection + residual -----------------
__global__ __launch_bounds__(256) void outproj_kernel(
    const _Float16* __restrict__ ao, const _Float16* __restrict__ WoT,
    const float* __restrict__ bo, const float* __restrict__ resid,
    float* __restrict__ out) {
  int lane = threadIdx.x & 31, wid = threadIdx.x >> 5;
  int gid = blockIdx.x * 8 + wid;        // 16384 waves total
  int ct = gid & 15;
  int rt = gid >> 4;                     // [0, 1024)
  int r0 = rt * 16, c0 = ct * 16;

  v8f acc = zero8();
#pragma unroll
  for (int kk = 0; kk < 8; ++kk) {
    v16h a = frag_rows(ao, r0, DM_, kk * 32, lane);
    v16h w = frag_rows(WoT, c0, DM_, kk * 32, lane);
    acc = wmma32(a, w, acc);
  }
  int nl = lane & 15, hf = lane >> 4;
#pragma unroll
  for (int v = 0; v < 8; ++v) {
    int r = r0 + v + 8 * hf;
    int c = c0 + nl;
    size_t idx = (size_t)r * DM_ + c;
    out[idx] = acc[v] + bo[c] + resid[idx];
  }
}

// ---------------- launch ----------------
extern "C" void kernel_launch(void* const* d_in, const int* in_sizes, int n_in,
                              void* d_out, int out_size, void* d_ws,
                              size_t ws_size, hipStream_t stream) {
  (void)in_sizes; (void)n_in; (void)out_size; (void)ws_size;
  const float* inputs  = (const float*)d_in[0];
  const float* pos_enc = (const float*)d_in[1];
  const float* gamma   = (const float*)d_in[2];
  const float* beta    = (const float*)d_in[3];
  const float* Wq      = (const float*)d_in[4];
  const float* bq      = (const float*)d_in[5];
  const float* Wk      = (const float*)d_in[6];
  const float* bk      = (const float*)d_in[7];
  const float* Wv      = (const float*)d_in[8];
  const float* bv      = (const float*)d_in[9];
  const float* Wp      = (const float*)d_in[10];
  const float* Wo      = (const float*)d_in[11];
  const float* bo      = (const float*)d_in[12];
  const float* pbu     = (const float*)d_in[13];
  const float* pbv     = (const float*)d_in[14];
  float* out = (float*)d_out;

  char* ws = (char*)d_ws;
  const size_t ACT = (size_t)B_ * T_ * DM_ * sizeof(_Float16);   // 8 MB
  const size_t WSZ = (size_t)DM_ * DM_ * sizeof(_Float16);       // 128 KB
  _Float16* xh  = (_Float16*)(ws);
  _Float16* ph  = (_Float16*)(ws + ACT);
  _Float16* WqT = (_Float16*)(ws + 2 * ACT);
  _Float16* WkT = (_Float16*)(ws + 2 * ACT + WSZ);
  _Float16* WvT = (_Float16*)(ws + 2 * ACT + 2 * WSZ);
  _Float16* WpT = (_Float16*)(ws + 2 * ACT + 3 * WSZ);
  _Float16* WoT = (_Float16*)(ws + 2 * ACT + 4 * WSZ);
  char* act2 = ws + 2 * ACT + 5 * WSZ;
  _Float16* qu  = (_Float16*)(act2);
  _Float16* qv  = (_Float16*)(act2 + ACT);
  _Float16* kh  = (_Float16*)(act2 + 2 * ACT);
  _Float16* vt  = (_Float16*)(act2 + 3 * ACT);
  _Float16* pph = (_Float16*)(act2 + 4 * ACT);
  _Float16* ao  = (_Float16*)(act2 + 5 * ACT);

  ln_kernel<<<B_ * T_, 256, 0, stream>>>(inputs, gamma, beta, xh);
  cvt_kernel<<<4096, 256, 0, stream>>>(pos_enc, ph, B_ * T_ * DM_);
  tr_kernel<<<DM_, 256, 0, stream>>>(Wq, WqT);
  tr_kernel<<<DM_, 256, 0, stream>>>(Wk, WkT);
  tr_kernel<<<DM_, 256, 0, stream>>>(Wv, WvT);
  tr_kernel<<<DM_, 256, 0, stream>>>(Wp, WpT);
  tr_kernel<<<DM_, 256, 0, stream>>>(Wo, WoT);

  // 4 mats x 16 batches x 64 row-tiles x 16 col-tiles = 65536 waves / 8 = 8192 blocks
  proj_kernel<<<8192, 256, 0, stream>>>(xh, ph, WqT, WkT, WvT, WpT, bq, bk, bv,
                                        pbu, pbv, qu, qv, kh, vt, pph);
  // 16 batches x 4 heads x 64 m-tiles = 4096 waves / 8 = 512 blocks
  attn_kernel<<<512, 256, 0, stream>>>(qu, qv, kh, vt, pph, ao);
  // 1024 row-tiles x 16 col-tiles = 16384 waves / 8 = 2048 blocks
  outproj_kernel<<<2048, 256, 0, stream>>>(ao, WoT, bo, inputs, out);
}
